// SRU_distill_3375844294676
// MI455X (gfx1250) — matrix operand order repaired and checked by reading, not compiled, not run on hardware
//
#include <hip/hip_runtime.h>

typedef unsigned short u16;
typedef __attribute__((ext_vector_type(16))) __bf16 v16bf;
typedef __attribute__((ext_vector_type(8)))  __bf16 v8bf;
typedef __attribute__((ext_vector_type(8)))  float  v8f;

#define NN   65535L        // real nodes
#define NPAD 65536L        // padded rows

// ---------------- workspace layout (bytes) ----------------
constexpr size_t O_XBF  = 0;                                  // NPAD*512 bf16
constexpr size_t O_HBF  = O_XBF  + NPAD*512*2;                // NPAD*512 bf16
constexpr size_t O_X32  = O_HBF  + NPAD*512*2;                // NPAD*512 f32
constexpr size_t O_XX   = O_X32  + NPAD*512*4;
constexpr size_t O_FF   = O_XX   + NPAD*512*4;
constexpr size_t O_RR   = O_FF   + NPAD*512*4;
constexpr size_t O_C    = O_RR   + NPAD*512*4;
constexpr size_t O_HID  = O_C    + NPAD*512*4;                // NPAD*256 f32
constexpr size_t O_WTF1 = O_HID  + NPAD*256*4;                // 512x96 bf16 (col-major WT)
constexpr size_t O_WTF2 = O_WTF1 + 512*96*2;                  // 512x512 bf16
constexpr size_t O_WTB  = O_WTF2 + 512*512*2;                 // 2560x512 bf16
constexpr size_t O_WTH  = O_WTB  + 2560*512*2;                // 1280x512 bf16
constexpr size_t O_BF1  = O_WTH  + 1280*512*2;                // 512 f32
constexpr size_t O_BF2  = O_BF1  + 512*4;
constexpr size_t O_BB   = O_BF2  + 512*4;                     // 2560 f32
constexpr size_t O_BH   = O_BB   + 2560*4;                    // 1280 f32

// ---------------- helpers ----------------
__device__ __forceinline__ u16 f2bf(float f) {
  unsigned u = __float_as_uint(f);
  u += 0x7FFFu + ((u >> 16) & 1u);          // round-to-nearest-even
  return (u16)(u >> 16);
}
__device__ __forceinline__ float sigm(float v) { return 1.f / (1.f + __expf(-v)); }

__device__ __forceinline__ v16bf cat16(v8bf lo, v8bf hi) {
  return __builtin_shufflevector(lo, hi, 0,1,2,3,4,5,6,7,8,9,10,11,12,13,14,15);
}
// A fragment (16x32 bf16) from LDS, ISA layout: lane&15 = row M, lane>>4 selects K octet
__device__ __forceinline__ v16bf load_afrag_lds(const u16* lds, int pitch, int lane, int k0) {
  int m = lane & 15, sel = lane >> 4;
  const u16* p = lds + m * pitch + k0 + sel * 8;
  v8bf a0 = *(const v8bf*)p;                // K = k0+8*sel .. +7
  v8bf a1 = *(const v8bf*)(p + 16);         // K = k0+16+8*sel .. +7
  return cat16(a0, a1);
}
// B fragment (32x16 bf16) from global col-major weights WT[n][K]
__device__ __forceinline__ v16bf load_bfrag(const u16* WT, int K, int colBase, int lane, int k0) {
  int n = lane & 15, sel = lane >> 4;
  const u16* p = WT + (size_t)(colBase + n) * K + k0 + sel * 8;
  v8bf b0 = *(const v8bf*)p;
  v8bf b1 = *(const v8bf*)(p + 16);
  return cat16(b0, b1);
}
__device__ __forceinline__ v8f wmma_bf16(v16bf a, v16bf b, v8f c) {
  return __builtin_amdgcn_wmma_f32_16x16x32_bf16(false, a, false, b, (short)0, c, false, false);
}

// ---------------- K0: weight repack to bf16 col-major + fused biases ----------------
struct PackArgs {
  const float *w_op, *w_tb, *w_ft, *w_jn;
  const float *w_op2, *w_tb2, *w_ft2, *w_jn2;
  const float *b_op, *b_tb, *b_ft, *b_jn;
  const float *b_op2, *b_tb2, *b_ft2, *b_jn2;
  const float *w_xdis, *w_xou, *w_hdis, *w_o1;
  const float *b_xdis, *b_xou, *b_hdis, *b_o1;
  u16 *WTf1, *WTf2, *WTbig, *WTh;
  float *Bf1, *Bf2, *Bbig, *Bh;
};
__global__ __launch_bounds__(256) void k_pack(PackArgs a) {
  long idx = (long)blockIdx.x * 256 + threadIdx.x;
  if (idx >= 2282240L) return;
  if (idx < 49152L) {                                   // WTf1: [512 cols][96 K], block diag
    int n = (int)(idx / 96), k = (int)(idx % 96);
    int b = n >> 7, lc = n & 127;
    const float* w[4] = {a.w_op, a.w_tb, a.w_ft, a.w_jn};
    const int K0[4] = {0, 4, 15, 33}, L[4] = {4, 11, 18, 37};
    float v = 0.f;
    if (k >= K0[b] && k < K0[b] + L[b]) v = w[b][(k - K0[b]) * 128 + lc];
    a.WTf1[idx] = f2bf(v);
  } else if (idx < 311296L) {                           // WTf2: [512][512] block diag
    long i = idx - 49152L; int n = (int)(i / 512), k = (int)(i % 512);
    int b = n >> 7;
    const float* w[4] = {a.w_op2, a.w_tb2, a.w_ft2, a.w_jn2};
    float v = ((k >> 7) == b) ? w[b][(k & 127) * 128 + (n & 127)] : 0.f;
    a.WTf2[i] = f2bf(v);
  } else if (idx < 1622016L) {                          // WTbig: [2560][512] = [xdis|xou]^T
    long i = idx - 311296L; int n = (int)(i / 512), k = (int)(i % 512);
    float v = (n < 1024) ? a.w_xdis[(size_t)k * 1024 + n] : a.w_xou[(size_t)k * 1536 + (n - 1024)];
    a.WTbig[i] = f2bf(v);
  } else if (idx < 2277376L) {                          // WTh: [1280][512] = [hdis|o1]^T
    long i = idx - 1622016L; int n = (int)(i / 512), k = (int)(i % 512);
    float v = (n < 1024) ? a.w_hdis[(size_t)k * 1024 + n] : a.w_o1[(size_t)k * 256 + (n - 1024)];
    a.WTh[i] = f2bf(v);
  } else if (idx < 2277888L) {
    int n = (int)(idx - 2277376L);
    const float* b[4] = {a.b_op, a.b_tb, a.b_ft, a.b_jn};
    a.Bf1[n] = b[n >> 7][n & 127];
  } else if (idx < 2278400L) {
    int n = (int)(idx - 2277888L);
    const float* b[4] = {a.b_op2, a.b_tb2, a.b_ft2, a.b_jn2};
    a.Bf2[n] = b[n >> 7][n & 127];
  } else if (idx < 2280960L) {
    int n = (int)(idx - 2278400L);
    a.Bbig[n] = (n < 1024) ? a.b_xdis[n] : a.b_xou[n - 1024];
  } else {
    int n = (int)(idx - 2280960L);
    a.Bh[n] = (n < 1024) ? a.b_hdis[n] : a.b_o1[n - 1024];
  }
}

// ---------------- K1: fused feature MLPs -> x (N x 512), two WMMA layers ----------------
// 16-node tiles; per wave: 4 independent column accumulators sharing one A fragment.
__global__ __launch_bounds__(256)
void k_features(const float* opf, const float* tbf, const float* ftf, const float* jnf,
                const u16* WTf1, const u16* WTf2, const float* Bf1, const float* Bf2,
                float* X32, u16* Xbf) {
  __shared__ __align__(16) u16 sF[16 * 104];   // padded features, K=96 (+pad)
  __shared__ __align__(16) u16 sH[16 * 536];   // layer-1 output, K=512 (+pad)
  const int t = threadIdx.x, lane = t & 31, wid = t >> 5;
  const long tile = blockIdx.x;
  for (int idx = t; idx < 16 * 104; idx += 256) {
    int r = idx / 104, c = idx % 104;
    long g = tile * 16 + r;
    float v = 0.f;
    if (g < NN) {
      if (c < 4)       v = opf[g * 4 + c];
      else if (c < 15) v = tbf[g * 11 + (c - 4)];
      else if (c < 33) v = ftf[g * 18 + (c - 15)];
      else if (c < 70) v = jnf[g * 37 + (c - 33)];
    }
    sF[r * 104 + c] = f2bf(v);
  }
  __syncthreads();
  const int n = lane & 15, sel = lane >> 4;
  const int cw = wid * 64;                              // this wave's 64-col span
  {                                                     // layer 1 (K=96)
    v8f acc[4] = {{}, {}, {}, {}};
    for (int ks = 0; ks < 3; ++ks) {
      v16bf a = load_afrag_lds(sF, 104, lane, ks * 32);
#pragma unroll
      for (int ct = 0; ct < 4; ++ct)
        acc[ct] = wmma_bf16(a, load_bfrag(WTf1, 96, cw + ct * 16, lane, ks * 32), acc[ct]);
    }
#pragma unroll
    for (int ct = 0; ct < 4; ++ct) {
      int col = cw + ct * 16 + n;
      float bias = Bf1[col];
#pragma unroll
      for (int j = 0; j < 8; ++j) {
        float v = acc[ct][j] + bias; v = v > 0.f ? v : 0.f;
        sH[(j + sel * 8) * 536 + col] = f2bf(v);
      }
    }
  }
  __syncthreads();
  {                                                     // layer 2 (K=512)
    v8f acc[4] = {{}, {}, {}, {}};
    for (int ks = 0; ks < 16; ++ks) {
      v16bf a = load_afrag_lds(sH, 536, lane, ks * 32);
#pragma unroll
      for (int ct = 0; ct < 4; ++ct)
        acc[ct] = wmma_bf16(a, load_bfrag(WTf2, 512, cw + ct * 16, lane, ks * 32), acc[ct]);
    }
#pragma unroll
    for (int ct = 0; ct < 4; ++ct) {
      int col = cw + ct * 16 + n;
      float bias = Bf2[col];
#pragma unroll
      for (int j = 0; j < 8; ++j) {
        long g = tile * 16 + j + sel * 8;
        float v = acc[ct][j] + bias; v = v > 0.f ? v : 0.f;
        X32[g * 512 + col] = v;
        Xbf[g * 512 + col] = f2bf(v);
      }
    }
  }
}

// ---------------- K2: x @ [w_xdis | w_xou]  (512 x 2560) ----------------
// 32-row tiles, each wave computes 32x32 (4 independent accumulators, 2x operand reuse).
__global__ __launch_bounds__(256)
void k_gemm_x(const u16* Xbf, const u16* WT, const float* B,
              float* out_xdis, float* XX, float* FF, float* RR) {
  __shared__ __align__(16) u16 sA[32 * 536];
  const int t = threadIdx.x, lane = t & 31, wid = t >> 5;
  const long tile = blockIdx.x;                         // 32-row tile
  const u16* src = Xbf + tile * 32 * 512;
  for (int idx = t; idx < 32 * 64; idx += 256) {
    int r = idx >> 6, ch = idx & 63;
    *(uint4*)&sA[r * 536 + ch * 8] = *(const uint4*)&src[r * 512 + ch * 8];
  }
  __syncthreads();
  const int n = lane & 15, sel = lane >> 4;
  const int cb = blockIdx.y * 256 + wid * 32;
  v8f a00 = {}, a01 = {}, a10 = {}, a11 = {};
  for (int ks = 0; ks < 16; ++ks) {
    v16bf af0 = load_afrag_lds(sA,            536, lane, ks * 32);
    v16bf af1 = load_afrag_lds(sA + 16 * 536, 536, lane, ks * 32);
    v16bf bf0 = load_bfrag(WT, 512, cb,      lane, ks * 32);
    v16bf bf1 = load_bfrag(WT, 512, cb + 16, lane, ks * 32);
    a00 = wmma_bf16(af0, bf0, a00);
    a01 = wmma_bf16(af0, bf1, a01);
    a10 = wmma_bf16(af1, bf0, a10);
    a11 = wmma_bf16(af1, bf1, a11);
  }
  auto epi = [&](v8f acc, int rowOff, int colOff) {
    int gc = cb + colOff + n;
    float bias = B[gc];
    if (gc < 1024) {                                      // x_dis -> d_out
#pragma unroll
      for (int j = 0; j < 8; ++j) {
        long g = tile * 32 + rowOff + j + sel * 8;
        if (g < NN) out_xdis[g * 1024 + gc] = acc[j] + bias;
      }
    } else if (gc < 1536) {                               // xx
      int c = gc - 1024;
#pragma unroll
      for (int j = 0; j < 8; ++j) { long g = tile * 32 + rowOff + j + sel * 8; XX[g * 512 + c] = acc[j] + bias; }
    } else if (gc < 2048) {                               // ff = sigmoid
      int c = gc - 1536;
#pragma unroll
      for (int j = 0; j < 8; ++j) { long g = tile * 32 + rowOff + j + sel * 8; FF[g * 512 + c] = sigm(acc[j] + bias); }
    } else {                                              // rr = sigmoid
      int c = gc - 2048;
#pragma unroll
      for (int j = 0; j < 8; ++j) { long g = tile * 32 + rowOff + j + sel * 8; RR[g * 512 + c] = sigm(acc[j] + bias); }
    }
  };
  epi(a00, 0, 0);  epi(a01, 0, 16);
  epi(a10, 16, 0); epi(a11, 16, 16);
}

// ---------------- K3: leaf init (node_order==0 => heap rows 32767..65535) ----------------
__global__ __launch_bounds__(256)
void k_leaves(const float* XX, const float* FF, const float* RR, const float* X32,
              float* C, u16* Hbf) {
  long idx = (long)blockIdx.x * 256 + threadIdx.x;
  if (idx >= 32769L * 128) return;
  long r = 32767 + idx / 128;
  int c4 = (int)(idx % 128) * 4;
  float4 xx = *(const float4*)&XX[r * 512 + c4];
  float4 ff = *(const float4*)&FF[r * 512 + c4];
  float4 rr = *(const float4*)&RR[r * 512 + c4];
  float4 x  = *(const float4*)&X32[r * 512 + c4];
  float4 cv, hv;
  cv.x = (1.f - ff.x) * xx.x;  cv.y = (1.f - ff.y) * xx.y;
  cv.z = (1.f - ff.z) * xx.z;  cv.w = (1.f - ff.w) * xx.w;
  hv.x = rr.x * tanhf(cv.x) + (1.f - rr.x) * x.x;
  hv.y = rr.y * tanhf(cv.y) + (1.f - rr.y) * x.y;
  hv.z = rr.z * tanhf(cv.z) + (1.f - rr.z) * x.z;
  hv.w = rr.w * tanhf(cv.w) + (1.f - rr.w) * x.w;
  *(float4*)&C[r * 512 + c4] = cv;
  ushort4 hb; hb.x = f2bf(hv.x); hb.y = f2bf(hv.y); hb.z = f2bf(hv.z); hb.w = f2bf(hv.w);
  *(ushort4*)&Hbf[r * 512 + c4] = hb;
}

// ---------------- K4: one tree level (children = 2i+1, 2i+2) ----------------
__global__ __launch_bounds__(256)
void k_level(const float* XX, const float* FF, const float* RR, const float* X32,
             float* C, u16* Hbf, long rowBase, long rows) {
  long idx = (long)blockIdx.x * 256 + threadIdx.x;
  if (idx >= rows * 128) return;
  long i = rowBase + idx / 128;
  int c4 = (int)(idx % 128) * 4;
  float4 cl = *(const float4*)&C[(2 * i + 1) * 512 + c4];
  float4 cr = *(const float4*)&C[(2 * i + 2) * 512 + c4];
  float4 xx = *(const float4*)&XX[i * 512 + c4];
  float4 ff = *(const float4*)&FF[i * 512 + c4];
  float4 rr = *(const float4*)&RR[i * 512 + c4];
  float4 x  = *(const float4*)&X32[i * 512 + c4];
  float4 cn, hv;
  cn.x = ff.x * (cl.x + cr.x) + (1.f - ff.x) * xx.x;
  cn.y = ff.y * (cl.y + cr.y) + (1.f - ff.y) * xx.y;
  cn.z = ff.z * (cl.z + cr.z) + (1.f - ff.z) * xx.z;
  cn.w = ff.w * (cl.w + cr.w) + (1.f - ff.w) * xx.w;
  hv.x = rr.x * tanhf(cn.x) + (1.f - rr.x) * x.x;
  hv.y = rr.y * tanhf(cn.y) + (1.f - rr.y) * x.y;
  hv.z = rr.z * tanhf(cn.z) + (1.f - rr.z) * x.z;
  hv.w = rr.w * tanhf(cn.w) + (1.f - rr.w) * x.w;
  *(float4*)&C[i * 512 + c4] = cn;
  ushort4 hb; hb.x = f2bf(hv.x); hb.y = f2bf(hv.y); hb.z = f2bf(hv.z); hb.w = f2bf(hv.w);
  *(ushort4*)&Hbf[i * 512 + c4] = hb;
}

// ---------------- K5: h @ [w_hdis | w_o1]  (512 x 1280), 32x32 per wave ----------------
__global__ __launch_bounds__(256)
void k_gemm_h(const u16* Hbf, const u16* WT, const float* B,
              float* out_hmdis, float* HID) {
  __shared__ __align__(16) u16 sA[32 * 536];
  const int t = threadIdx.x, lane = t & 31, wid = t >> 5;
  const long tile = blockIdx.x;
  const u16* src = Hbf + tile * 32 * 512;
  for (int idx = t; idx < 32 * 64; idx += 256) {
    int r = idx >> 6, ch = idx & 63;
    *(uint4*)&sA[r * 536 + ch * 8] = *(const uint4*)&src[r * 512 + ch * 8];
  }
  __syncthreads();
  const int n = lane & 15, sel = lane >> 4;
  const int cb = blockIdx.y * 256 + wid * 32;
  v8f a00 = {}, a01 = {}, a10 = {}, a11 = {};
  for (int ks = 0; ks < 16; ++ks) {
    v16bf af0 = load_afrag_lds(sA,            536, lane, ks * 32);
    v16bf af1 = load_afrag_lds(sA + 16 * 536, 536, lane, ks * 32);
    v16bf bf0 = load_bfrag(WT, 512, cb,      lane, ks * 32);
    v16bf bf1 = load_bfrag(WT, 512, cb + 16, lane, ks * 32);
    a00 = wmma_bf16(af0, bf0, a00);
    a01 = wmma_bf16(af0, bf1, a01);
    a10 = wmma_bf16(af1, bf0, a10);
    a11 = wmma_bf16(af1, bf1, a11);
  }
  auto epi = [&](v8f acc, int rowOff, int colOff) {
    int gc = cb + colOff + n;
    float bias = B[gc];
    if (gc < 1024) {                                      // hm_dis -> d_out
#pragma unroll
      for (int j = 0; j < 8; ++j) {
        long g = tile * 32 + rowOff + j + sel * 8;
        if (g < NN) out_hmdis[g * 1024 + gc] = acc[j] + bias;
      }
    } else {                                              // hid = relu
      int c = gc - 1024;
#pragma unroll
      for (int j = 0; j < 8; ++j) {
        long g = tile * 32 + rowOff + j + sel * 8;
        float v = acc[j] + bias;
        HID[g * 256 + c] = v > 0.f ? v : 0.f;
      }
    }
  };
  epi(a00, 0, 0);  epi(a01, 0, 16);
  epi(a10, 16, 0); epi(a11, 16, 16);
}

// ---------------- K6: logits + sigmoid (one wave per node) ----------------
__global__ __launch_bounds__(256)
void k_final(const float* HID, const float* w_o2, const float* b_o2, float* dout) {
  const int lane = threadIdx.x & 31, wid = threadIdx.x >> 5;
  long node = (long)blockIdx.x * 8 + wid;
  if (node >= NN) return;
  const float4* h = (const float4*)&HID[node * 256 + lane * 8];
  const float4* w = (const float4*)&w_o2[lane * 8];
  float4 h0 = h[0], h1 = h[1], w0 = w[0], w1 = w[1];
  float s = h0.x * w0.x + h0.y * w0.y + h0.z * w0.z + h0.w * w0.w
          + h1.x * w1.x + h1.y * w1.y + h1.z * w1.z + h1.w * w1.w;
  for (int off = 16; off > 0; off >>= 1) s += __shfl_xor(s, off, 32);
  if (lane == 0) {
    float logit = s + b_o2[0];
    dout[NN + node] = logit;        // logits
    dout[node] = sigm(logit);       // out
  }
}

// ---------------- host launcher ----------------
extern "C" void kernel_launch(void* const* d_in, const int* in_sizes, int n_in,
                              void* d_out, int out_size, void* d_ws, size_t ws_size,
                              hipStream_t stream) {
  (void)in_sizes; (void)n_in; (void)out_size; (void)ws_size;
  char* ws = (char*)d_ws;
  u16*   Xbf  = (u16*)  (ws + O_XBF);
  u16*   Hbf  = (u16*)  (ws + O_HBF);
  float* X32  = (float*)(ws + O_X32);
  float* XX   = (float*)(ws + O_XX);
  float* FF   = (float*)(ws + O_FF);
  float* RR   = (float*)(ws + O_RR);
  float* C    = (float*)(ws + O_C);
  float* HID  = (float*)(ws + O_HID);
  u16*   WTf1 = (u16*)  (ws + O_WTF1);
  u16*   WTf2 = (u16*)  (ws + O_WTF2);
  u16*   WTb  = (u16*)  (ws + O_WTB);
  u16*   WTh  = (u16*)  (ws + O_WTH);
  float* Bf1  = (float*)(ws + O_BF1);
  float* Bf2  = (float*)(ws + O_BF2);
  float* Bb   = (float*)(ws + O_BB);
  float* Bh   = (float*)(ws + O_BH);
  float* dout = (float*)d_out;

  PackArgs pa;
  pa.w_op  = (const float*)d_in[7];  pa.b_op  = (const float*)d_in[8];
  pa.w_op2 = (const float*)d_in[9];  pa.b_op2 = (const float*)d_in[10];
  pa.w_tb  = (const float*)d_in[11]; pa.b_tb  = (const float*)d_in[12];
  pa.w_tb2 = (const float*)d_in[13]; pa.b_tb2 = (const float*)d_in[14];
  pa.w_ft  = (const float*)d_in[15]; pa.b_ft  = (const float*)d_in[16];
  pa.w_ft2 = (const float*)d_in[17]; pa.b_ft2 = (const float*)d_in[18];
  pa.w_jn  = (const float*)d_in[19]; pa.b_jn  = (const float*)d_in[20];
  pa.w_jn2 = (const float*)d_in[21]; pa.b_jn2 = (const float*)d_in[22];
  pa.w_xdis = (const float*)d_in[23]; pa.b_xdis = (const float*)d_in[24];
  pa.w_hdis = (const float*)d_in[25]; pa.b_hdis = (const float*)d_in[26];
  pa.w_xou  = (const float*)d_in[27]; pa.b_xou  = (const float*)d_in[28];
  pa.w_o1   = (const float*)d_in[29]; pa.b_o1   = (const float*)d_in[30];
  pa.WTf1 = WTf1; pa.WTf2 = WTf2; pa.WTbig = WTb; pa.WTh = WTh;
  pa.Bf1 = Bf1; pa.Bf2 = Bf2; pa.Bbig = Bb; pa.Bh = Bh;
  k_pack<<<8916, 256, 0, stream>>>(pa);

  k_features<<<4096, 256, 0, stream>>>((const float*)d_in[0], (const float*)d_in[1],
                                       (const float*)d_in[2], (const float*)d_in[3],
                                       WTf1, WTf2, Bf1, Bf2, X32, Xbf);

  k_gemm_x<<<dim3(2048, 10), 256, 0, stream>>>(Xbf, WTb, Bb, dout + 2 * NN, XX, FF, RR);

  k_leaves<<<16385, 256, 0, stream>>>(XX, FF, RR, X32, C, Hbf);

  for (int d = 14; d >= 0; --d) {
    long rows = 1L << d;
    long work = rows * 128;
    int blocks = (int)((work + 255) / 256); if (blocks < 1) blocks = 1;
    k_level<<<blocks, 256, 0, stream>>>(XX, FF, RR, X32, C, Hbf, rows - 1, rows);
  }

  k_gemm_h<<<dim3(2048, 5), 256, 0, stream>>>(Hbf, WTh, Bh, dout + 2 * NN + 1024 * NN, HID);

  k_final<<<8192, 256, 0, stream>>>(HID, (const float*)d_in[31], (const float*)d_in[32], dout);
}
